// Attention_41558103556618
// MI455X (gfx1250) — matrix operand order repaired and checked by reading, compiled, not run on hardware
//
#include <hip/hip_runtime.h>
#include <hip/hip_bf16.h>
#include <stdint.h>

typedef __attribute__((ext_vector_type(16))) _Float16 v16h;
typedef __attribute__((ext_vector_type(8)))  _Float16 v8h;
typedef __attribute__((ext_vector_type(8)))  float    v8f;
typedef __attribute__((ext_vector_type(4)))  uint32_t v4u;
typedef __attribute__((ext_vector_type(8)))  int      v8i_;
typedef __attribute__((ext_vector_type(4)))  int      v4i_;

#define WMMA16(a, b, c) \
    __builtin_amdgcn_wmma_f32_16x16x32_f16(false, (a), false, (b), (short)0, (c), false, false)

constexpr int Bb = 2, Tt = 2048, Ee = 1024, Hh = 16, Dd = 64;
constexpr int S  = Bb * Tt;      // 4096 token rows
constexpr int N3 = 3 * Ee;       // 3072

// LDS tile row stride: 32 halves (64B) + 16B TDM padding = 80B = 40 halves
constexpr int ROWH = 40;

static __device__ inline v16h combine8(v8h lo, v8h hi) {
    v16h r;
#pragma unroll
    for (int i = 0; i < 8; ++i) { r[i] = lo[i]; r[i + 8] = hi[i]; }
    return r;
}

static __device__ inline float rmax16(float v) {
    v = fmaxf(v, __shfl_xor(v, 1, 32));
    v = fmaxf(v, __shfl_xor(v, 2, 32));
    v = fmaxf(v, __shfl_xor(v, 4, 32));
    v = fmaxf(v, __shfl_xor(v, 8, 32));
    return v;
}
static __device__ inline float rsum16(float v) {
    v += __shfl_xor(v, 1, 32);
    v += __shfl_xor(v, 2, 32);
    v += __shfl_xor(v, 4, 32);
    v += __shfl_xor(v, 8, 32);
    return v;
}

// ---- Tensor Data Mover: async 2D tile (rows x 32 halves) global -> LDS ------
// D# per CDNA5 ISA. data_size=2B, pad every 16 DWORDs
// (64B = one 32-half row) by 4 DWORDs (16B) -> 80B LDS row stride.
static __device__ inline void tdm_load_tile(uint32_t lds_off, const _Float16* gsrc,
                                            uint32_t ktotal, uint32_t ntotal,
                                            uint32_t tile_rows) {
    uint64_t ga = (uint64_t)(uintptr_t)gsrc;
    v4u g0;
    g0[0] = 1u;                                           // count=1 valid descriptor
    g0[1] = lds_off;                                      // LDS byte address
    g0[2] = (uint32_t)ga;                                 // global_addr[31:0]
    g0[3] = ((uint32_t)(ga >> 32) & 0x01FFFFFFu) | 0x80000000u;  // addr[56:32] | type=2
    v8i_ g1;
    g1[0] = (int)((1u << 16)      // data_size = 2 bytes
                | (1u << 20)      // pad_enable
                | (3u << 22)      // pad_interval: 16 DWORDs (one 64B row)
                | (3u << 25));    // pad_amount: 4 DWORDs (16B)
    g1[1] = (int)((ktotal & 0xFFFFu) << 16);              // tensor_dim0[15:0] @ bit48
    g1[2] = (int)((ktotal >> 16) | ((ntotal & 0xFFFFu) << 16));  // dim0 hi | dim1 lo
    g1[3] = (int)((ntotal >> 16) | (32u << 16));          // dim1 hi | tile_dim0=32
    g1[4] = (int)tile_rows;                               // tile_dim1
    g1[5] = (int)ktotal;                                  // tensor_dim0_stride lo32
    g1[6] = 0;
    g1[7] = 0;
    v4i_ gz = {0, 0, 0, 0};
#if __clang_major__ >= 23
    v8i_ gz8 = {0, 0, 0, 0, 0, 0, 0, 0};
    __builtin_amdgcn_tensor_load_to_lds(g0, g1, gz, gz, gz8, 0);
#else
    __builtin_amdgcn_tensor_load_to_lds(g0, g1, gz, gz, 0);
#endif
}

// ---------------- prep: f32 -> f16 convert / convert+transpose ----------------
__global__ void cvt_f16_kernel(const float* __restrict__ src, _Float16* __restrict__ dst, int n) {
    for (int i = blockIdx.x * blockDim.x + threadIdx.x; i < n; i += gridDim.x * blockDim.x)
        dst[i] = (_Float16)src[i];
}

// w: [K][N] row-major -> wt: [N][K] row-major (f16)
__global__ void cvt_tr_kernel(const float* __restrict__ w, _Float16* __restrict__ wt, int K, int N) {
    int total = K * N;
    for (int i = blockIdx.x * blockDim.x + threadIdx.x; i < total; i += gridDim.x * blockDim.x) {
        int k = i / N, n = i - k * N;
        wt[(size_t)n * K + k] = (_Float16)w[i];
    }
}

// ---- shared TDM-fed GEMM core: block tile M=64 x N=256, 16 waves ------------
// acc[t] = (A[m0w.., :] @ B^T[n0+64*nsub+16t.., :]) accumulated over K=Ee.
// shA: [2][64*ROWH] halves, shB: [2][256*ROWH] halves.
static __device__ inline void gemm_core_tdm(const _Float16* __restrict__ aglob,
                                            const _Float16* __restrict__ bglob,
                                            int m0, int n0, int ntotal,
                                            _Float16 (*shA)[64 * ROWH],
                                            _Float16 (*shB)[256 * ROWH],
                                            v8f acc[4]) {
    const int lane = threadIdx.x & 31;
    const int wave = threadIdx.x >> 5;
    const int hi   = lane >> 4;
    const int l15  = lane & 15;
    const int msub = wave & 3;          // 4 x 16 rows
    const int nsub = wave >> 2;         // 4 x 64 cols

    uint32_t offA[2], offB[2];
    offA[0] = (uint32_t)(uintptr_t)&shA[0][0];
    offA[1] = (uint32_t)(uintptr_t)&shA[1][0];
    offB[0] = (uint32_t)(uintptr_t)&shB[0][0];
    offB[1] = (uint32_t)(uintptr_t)&shB[1][0];

    if (wave == 0) {
        tdm_load_tile(offA[0], aglob + (size_t)m0 * Ee, Ee, S, 64);
        tdm_load_tile(offB[0], bglob + (size_t)n0 * Ee, Ee, (uint32_t)ntotal, 256);
    }

    for (int k0 = 0; k0 < Ee; k0 += 32) {
        const int cur = (k0 >> 5) & 1;
        const int nxt = cur ^ 1;
        if (wave == 0) __builtin_amdgcn_s_wait_tensorcnt(0);
        __syncthreads();   // buf[cur] filled; buf[nxt] fully consumed last iter
        if (wave == 0 && k0 + 32 < Ee) {
            tdm_load_tile(offA[nxt], aglob + (size_t)m0 * Ee + k0 + 32, Ee, S, 64);
            tdm_load_tile(offB[nxt], bglob + (size_t)n0 * Ee + k0 + 32, Ee, (uint32_t)ntotal, 256);
        }
        const _Float16* arow = &shA[cur][(16 * msub + l15) * ROWH];
        v16h a = combine8(*(const v8h*)(arow + hi * 8),
                          *(const v8h*)(arow + 16 + hi * 8));
#pragma unroll
        for (int t = 0; t < 4; ++t) {
            const _Float16* brow = &shB[cur][(64 * nsub + 16 * t + l15) * ROWH];
            v16h b = combine8(*(const v8h*)(brow + hi * 8),
                              *(const v8h*)(brow + 16 + hi * 8));
            acc[t] = WMMA16(a, b, acc[t]);
        }
    }
}

// ---------------- QKV GEMM: xh[S,E] @ wqkvT[N3,E]^T + bias -> scatter q,k,vT --
__global__ __launch_bounds__(512) void qkv_gemm_kernel(
    const _Float16* __restrict__ xh, const _Float16* __restrict__ wt,
    const float* __restrict__ bias,
    _Float16* __restrict__ qb, _Float16* __restrict__ kb, _Float16* __restrict__ vt) {
    __shared__ _Float16 shA[2][64 * ROWH];
    __shared__ _Float16 shB[2][256 * ROWH];

    const int lane = threadIdx.x & 31;
    const int wave = threadIdx.x >> 5;
    const int hi   = lane >> 4;
    const int l15  = lane & 15;
    const int msub = wave & 3;
    const int nsub = wave >> 2;
    const int m0   = blockIdx.x * 64;
    const int n0   = blockIdx.y * 256;

    v8f acc[4];
#pragma unroll
    for (int t = 0; t < 4; ++t) {
        float bv = bias[n0 + 64 * nsub + 16 * t + l15];
#pragma unroll
        for (int g = 0; g < 8; ++g) acc[t][g] = bv;
    }

    gemm_core_tdm(xh, wt, m0, n0, N3, shA, shB, acc);

#pragma unroll
    for (int t = 0; t < 4; ++t) {
#pragma unroll
        for (int g = 0; g < 8; ++g) {
            int row  = m0 + 16 * msub + g + 8 * hi;      // token row in [0,S)
            int col  = n0 + 64 * nsub + 16 * t + l15;    // column in [0,3E)
            int sect = col >> 10;                        // 0=q 1=k 2=v
            int e    = col & 1023;
            int hh   = e >> 6;
            int dd   = e & 63;
            int bi   = row >> 11;
            int tok  = row & 2047;
            _Float16 val = (_Float16)acc[t][g];
            if (sect == 0)
                qb[(((size_t)(bi * Hh + hh) * Tt + tok) << 6) + dd] = val;
            else if (sect == 1)
                kb[(((size_t)(bi * Hh + hh) * Tt + tok) << 6) + dd] = val;
            else
                vt[(((size_t)(bi * Hh + hh) * Dd + dd) << 11) + tok] = val;
        }
    }
}

// ---------------- flash attention: one wave per 16-query tile ----------------
__global__ __launch_bounds__(128) void attn_kernel(
    const _Float16* __restrict__ qb, const _Float16* __restrict__ kb,
    const _Float16* __restrict__ vt, _Float16* __restrict__ yh) {
    __shared__ _Float16 psh[4][16 * 32];

    const int lane = threadIdx.x & 31;
    const int wave = threadIdx.x >> 5;
    const int hi   = lane >> 4;
    const int l15  = lane & 15;

    const int tile = blockIdx.x * 4 + wave;      // [0, B*H*T/16)
    const int qt   = tile & (Tt / 16 - 1);
    const int bh   = tile / (Tt / 16);
    const int q0   = qt * 16;

    const _Float16* qbase = qb + (size_t)bh * Tt * Dd;
    const _Float16* kbase = kb + (size_t)bh * Tt * Dd;
    const _Float16* vbase = vt + (size_t)bh * Dd * Tt;

    const _Float16* qrow = qbase + (size_t)(q0 + l15) * Dd;
    v16h a0 = combine8(*(const v8h*)(qrow + hi * 16),      *(const v8h*)(qrow + 8 + hi * 16));
    v16h a1 = combine8(*(const v8h*)(qrow + 32 + hi * 16), *(const v8h*)(qrow + 40 + hi * 16));

    v8f o[4] = {v8f{}, v8f{}, v8f{}, v8f{}};
    float mrow[8], lrow[8];
#pragma unroll
    for (int g = 0; g < 8; ++g) { mrow[g] = -1e30f; lrow[g] = 0.0f; }

    const float sc = 0.125f;                     // 1/sqrt(64)
    const int kend = q0 + 16;                    // causal: keys <= last query

    for (int k0 = 0; k0 < kend; k0 += 32) {
        // ---- S = Q K^T for 32 keys (two 16-wide column sub-tiles) ----
        v8f s0 = {}, s1 = {};
        const _Float16* kr0 = kbase + (size_t)(k0 + l15) * Dd;
        s0 = WMMA16(a0, *(const v16h*)(kr0 + hi * 16), s0);
        s0 = WMMA16(a1, *(const v16h*)(kr0 + 32 + hi * 16), s0);
        const _Float16* kr1 = kbase + (size_t)(k0 + 16 + l15) * Dd;
        s1 = WMMA16(a0, *(const v16h*)(kr1 + hi * 16), s1);
        s1 = WMMA16(a1, *(const v16h*)(kr1 + 32 + hi * 16), s1);

        const int key0 = k0 + l15;
        const int key1 = k0 + 16 + l15;
        float p0[8], p1[8];
#pragma unroll
        for (int g = 0; g < 8; ++g) {
            int qi = q0 + g + 8 * hi;
            float x0 = (key0 > qi) ? -1e30f : s0[g] * sc;
            float x1 = (key1 > qi) ? -1e30f : s1[g] * sc;
            float bm = rmax16(fmaxf(x0, x1));
            float mn = fmaxf(mrow[g], bm);
            float resc = __expf(mrow[g] - mn);
            x0 = __expf(x0 - mn);
            x1 = __expf(x1 - mn);
            float rs = rsum16(x0 + x1);
            lrow[g] = lrow[g] * resc + rs;
            mrow[g] = mn;
            p0[g] = x0;
            p1[g] = x1;
#pragma unroll
            for (int t = 0; t < 4; ++t) o[t][g] *= resc;
        }

        // ---- C-layout P -> A-layout fragment via LDS (in-wave ordering) ----
        _Float16* pp = psh[wave];
#pragma unroll
        for (int g = 0; g < 8; ++g) {
            int r = g + 8 * hi;
            pp[r * 32 + l15]      = (_Float16)p0[g];
            pp[r * 32 + 16 + l15] = (_Float16)p1[g];
        }
        asm volatile("" ::: "memory");
        const _Float16* prow = pp + l15 * 32;
        v16h pa = combine8(*(const v8h*)(prow + hi * 16),
                           *(const v8h*)(prow + 8 + hi * 16));

        // ---- O += P V over this 32-key block (4 d-tiles of 16) ----
#pragma unroll
        for (int t = 0; t < 4; ++t) {
            const _Float16* vp = vbase + (size_t)(16 * t + l15) * Tt + k0 + hi * 16;
            o[t] = WMMA16(pa, *(const v16h*)vp, o[t]);
        }
        asm volatile("" ::: "memory");
    }

    // ---- normalize and write y as f16 [S, E] row-major ----
    const int bi = bh >> 4;
    const int hh = bh & 15;
    float inv[8];
#pragma unroll
    for (int g = 0; g < 8; ++g) inv[g] = 1.0f / lrow[g];
#pragma unroll
    for (int t = 0; t < 4; ++t) {
#pragma unroll
        for (int g = 0; g < 8; ++g) {
            int tok = q0 + g + 8 * hi;
            yh[(size_t)(bi * Tt + tok) * Ee + hh * 64 + 16 * t + l15] =
                (_Float16)(o[t][g] * inv[g]);
        }
    }
}

// ---------------- output projection: yh[S,E] @ wprojT[E,E]^T + bias -> f32 ---
__global__ __launch_bounds__(512) void proj_gemm_kernel(
    const _Float16* __restrict__ yh, const _Float16* __restrict__ wt,
    const float* __restrict__ bias, float* __restrict__ out) {
    __shared__ _Float16 shA[2][64 * ROWH];
    __shared__ _Float16 shB[2][256 * ROWH];

    const int lane = threadIdx.x & 31;
    const int wave = threadIdx.x >> 5;
    const int hi   = lane >> 4;
    const int l15  = lane & 15;
    const int msub = wave & 3;
    const int nsub = wave >> 2;
    const int m0   = blockIdx.x * 64;
    const int n0   = blockIdx.y * 256;

    v8f acc[4];
#pragma unroll
    for (int t = 0; t < 4; ++t) {
        float bv = bias[n0 + 64 * nsub + 16 * t + l15];
#pragma unroll
        for (int g = 0; g < 8; ++g) acc[t][g] = bv;
    }

    gemm_core_tdm(yh, wt, m0, n0, Ee, shA, shB, acc);

#pragma unroll
    for (int t = 0; t < 4; ++t)
#pragma unroll
        for (int g = 0; g < 8; ++g)
            out[(size_t)(m0 + 16 * msub + g + 8 * hi) * Ee + n0 + 64 * nsub + 16 * t + l15] =
                acc[t][g];
}

// -----------------------------------------------------------------------------
extern "C" void kernel_launch(void* const* d_in, const int* in_sizes, int n_in,
                              void* d_out, int out_size, void* d_ws, size_t ws_size,
                              hipStream_t stream) {
    (void)in_sizes; (void)n_in; (void)out_size; (void)ws_size;
    const float* x      = (const float*)d_in[0];   // [B,T,E]
    const float* w_qkv  = (const float*)d_in[1];   // [E,3E]
    const float* b_qkv  = (const float*)d_in[2];   // [3E]
    const float* w_proj = (const float*)d_in[3];   // [E,E]
    const float* b_proj = (const float*)d_in[4];   // [E]
    float* out = (float*)d_out;                    // [B,T,E] f32

    // workspace carve (bytes): all regions are 256B-aligned multiples
    uint8_t* w = (uint8_t*)d_ws;
    _Float16* xh     = (_Float16*)(w);                        // S*E       = 8 MB
    _Float16* wqkvT  = (_Float16*)(w + (8u << 20));           // N3*E      = 6 MB
    _Float16* wprojT = (_Float16*)(w + (14u << 20));          // E*E       = 2 MB
    _Float16* qb     = (_Float16*)(w + (16u << 20));          // B*H*T*D   = 8 MB
    _Float16* kb     = (_Float16*)(w + (24u << 20));          // B*H*T*D   = 8 MB
    _Float16* vt     = (_Float16*)(w + (32u << 20));          // B*H*D*T   = 8 MB
    _Float16* yhb    = (_Float16*)(w + (40u << 20));          // S*E       = 8 MB

    cvt_f16_kernel<<<4096, 256, 0, stream>>>(x, xh, S * Ee);
    cvt_tr_kernel<<<4096, 256, 0, stream>>>(w_qkv, wqkvT, Ee, N3);
    cvt_tr_kernel<<<2048, 256, 0, stream>>>(w_proj, wprojT, Ee, Ee);

    // QKV GEMM: M blocks = 4096/64 = 64, N blocks = 3072/256 = 12
    qkv_gemm_kernel<<<dim3(S / 64, N3 / 256), 512, 0, stream>>>(
        xh, wqkvT, b_qkv, qb, kb, vt);

    // attention: B*H*(T/16) = 4096 wave-tiles, 4 per block
    attn_kernel<<<(Bb * Hh * (Tt / 16)) / 4, 128, 0, stream>>>(qb, kb, vt, yhb);

    // projection: M blocks = 64, N blocks = 1024/256 = 4
    proj_gemm_kernel<<<dim3(S / 64, Ee / 256), 512, 0, stream>>>(
        yhb, wprojT, b_proj, out);
}
